// AugmentedFlowLayer_14044543058372
// MI455X (gfx1250) — matrix operand order repaired and checked by reading, not compiled, not run on hardware
//
#include <hip/hip_runtime.h>
#include <hip/hip_bf16.h>

#define NREAL 257
#define NP    272
#define HID   64
#define HEADS 4
#define COEF  256
#define RBF_N 50

typedef __attribute__((ext_vector_type(16))) _Float16 v16h;
typedef __attribute__((ext_vector_type(8)))  float    v8f;

__device__ __forceinline__ v8f wmma_f16(v16h a, v16h b, v8f c) {
  // D = A(16x32 f16) * B(32x16 f16) + C(16x16 f32)
  return __builtin_amdgcn_wmma_f32_16x16x32_f16(false, a, false, b, (short)0, c, false, false);
}

__device__ __forceinline__ v8f zero_v8f() {
  v8f z;
#pragma unroll
  for (int e = 0; e < 8; ++e) z[e] = 0.0f;
  return z;
}

__device__ __forceinline__ v16h load_b_frag(const _Float16* pb, int lane) {
  // packed fragment: 32 lanes x 16 halves, contiguous 32B per lane
  return *(const v16h*)(pb + lane * 16);
}

__device__ __forceinline__ float silu_f(float x) { return x / (1.0f + __expf(-x)); }

__device__ __forceinline__ float rbf_val(float d, int r) {
  const float start = 0.006737946999085467f;                 // exp(-5)
  const float width = (2.0f / RBF_N) * (1.0f - start);
  const float beta  = 1.0f / (width * width);
  float mean = start + (1.0f - start) * ((float)r / (float)(RBF_N - 1));
  float cut  = (d < 5.0f) ? (0.5f * (__cosf(d * 0.6283185307179586f) + 1.0f)) : 0.0f;
  float t = __expf(-d) - mean;
  return cut * __expf(-beta * t * t);
}

struct LayerPtrs {
  const float *ei_b, *eo1_b, *eo2_b, *log_gamma;
  const float *n1_b, *n1_w, *n2_b, *n2_w;
  const float *p1_b, *p1_w, *p2_b, *p2_w;
  const float *sem_b, *sem_w;
  const float *v1_b, *v1_w, *v2_w;
  const _Float16 *ei_p, *eo1_p, *eo2_p, *mix_p;   // packed WMMA B fragments
};

// ---------------------------------------------------------------------------
// Pack a (K x N) f32 weight into WMMA B-fragment layout, padded with zeros.
// out[(nt*ksteps+ks)*512 + lane*16 + e] = W[ks*32+(lane>>4)*8+(e<8?e:e+8)][nt*16+(lane&15)]
// ---------------------------------------------------------------------------
__global__ void pack_weight(const float* __restrict__ w, _Float16* __restrict__ out,
                            int K, int N, int ksteps, int ntiles) {
  int total = ntiles * ksteps * 512;
  for (int idx = blockIdx.x * 256 + threadIdx.x; idx < total; idx += gridDim.x * 256) {
    int frag = idx >> 9, within = idx & 511;
    int lane = within >> 4, e = within & 15;
    int ks = frag % ksteps, nt = frag / ksteps;
    int kg = ks * 32 + (lane >> 4) * 8 + (e < 8 ? e : e + 8);
    int ng = nt * 16 + (lane & 15);
    out[idx] = (_Float16)((kg < K && ng < N) ? w[kg * N + ng] : 0.0f);
  }
}

// ---------------------------------------------------------------------------
// prep: hh = [h | |x|^2], zero row 256 and pads; h0 = emb_in(hh); x/v padded
// ---------------------------------------------------------------------------
__global__ __launch_bounds__(64) void prep_kernel(
    const float* __restrict__ h_in, const float* __restrict__ x_in,
    const float* __restrict__ emb_b, const float* __restrict__ emb_w,
    float* __restrict__ h0, float* __restrict__ xp, float* __restrict__ vp) {
  __shared__ float hh[17];
  int i = blockIdx.x, tid = threadIdx.x;
  if (tid < 16) hh[tid] = (i < 256) ? h_in[i * 16 + tid] : 0.0f;
  if (tid == 16) {
    float s = 0.0f;
    if (i < 256) {
      float a = x_in[i * 3 + 0], b = x_in[i * 3 + 1], c = x_in[i * 3 + 2];
      s = a * a + b * b + c * c;
    }
    hh[16] = s;
  }
  __syncthreads();
  float s = emb_b[tid];
#pragma unroll
  for (int k = 0; k < 17; ++k) s += hh[k] * emb_w[k * 64 + tid];
  h0[i * 64 + tid] = s;
  if (tid < 3) {
    xp[i * 3 + tid] = (i < 256) ? x_in[i * 3 + tid] : 0.0f;
    vp[i * 3 + tid] = 0.0f;
  }
}

// f32 -> f16 node features; zero pad rows (keeps NaNs out of the WMMA chain)
__global__ void h_to_f16(const float* __restrict__ h, _Float16* __restrict__ h16) {
  int idx = blockIdx.x * 256 + threadIdx.x;
  if (idx < NP * 64) {
    int row = idx >> 6;
    h16[idx] = (_Float16)((row < NREAL) ? h[idx] : 0.0f);
  }
}

// ---------------------------------------------------------------------------
// pass1: per (itile,jtile) edge tile, per-wave 48-WMMA chain producing
// h_e (f16, global) and combined attention logits (celu(sem) - d*gamma, masked)
// ---------------------------------------------------------------------------
__global__ __launch_bounds__(256) void edge_pass1(
    const _Float16* __restrict__ h16, const float* __restrict__ x,
    LayerPtrs P, _Float16* __restrict__ he, float* __restrict__ logits) {
  __shared__ _Float16 sh_i[16 * 64];
  __shared__ _Float16 sh_j[16 * 64];
  __shared__ float sd[256];
  __shared__ _Float16 stage[8][16 * 64];

  int itile = blockIdx.x, jtile = blockIdx.y;
  int tid = threadIdx.x, wave = tid >> 5, lane = tid & 31;

  for (int t = tid; t < 1024; t += 256) {
    sh_i[t] = h16[itile * 1024 + t];
    sh_j[t] = h16[jtile * 1024 + t];
  }
  {
    int il = tid >> 4, jl = tid & 15;
    int gi = itile * 16 + il, gj = jtile * 16 + jl;
    float dx = x[gj * 3 + 0] - x[gi * 3 + 0];
    float dy = x[gj * 3 + 1] - x[gi * 3 + 1];
    float dz = x[gj * 3 + 2] - x[gi * 3 + 2];
    sd[tid] = sqrtf(dx * dx + dy * dy + dz * dz + 1e-14f);
  }
  __syncthreads();

  _Float16* st = stage[wave];
  const int m = lane & 15;            // A-matrix row (edge j-local)
  const int koff = (lane >> 4) * 8;   // A-matrix K offset per half-wave
  const int n = lane & 15;            // C-matrix column
  const int mhi = (lane >> 4) * 8;    // C-matrix row offset

  for (int g = wave; g < 16; g += 8) {   // g = i-local within tile
    int gi = itile * 16 + g;
    // ---- GEMM1: h_cat(128) @ W_edge_in -> t1 (16 x 64)
    v8f acc[4];
#pragma unroll
    for (int q = 0; q < 4; ++q) acc[q] = zero_v8f();
#pragma unroll
    for (int ks = 0; ks < 4; ++ks) {
      v16h a;
#pragma unroll
      for (int e = 0; e < 16; ++e) {
        int k = ks * 32 + koff + (e < 8 ? e : e + 8);
        a[e] = (k < 64) ? sh_j[m * 64 + k] : sh_i[g * 64 + (k - 64)];
      }
#pragma unroll
      for (int nt = 0; nt < 4; ++nt)
        acc[nt] = wmma_f16(a, load_b_frag(P.ei_p + (nt * 4 + ks) * 512, lane), acc[nt]);
    }
    // filt staging: cols [0..49]=rbf(d)*(t1+b), 50=d, 51..63=0
#pragma unroll
    for (int nt = 0; nt < 4; ++nt) {
      int col = nt * 16 + n;
#pragma unroll
      for (int r = 0; r < 8; ++r) {
        int jl = mhi + r;
        float d = sd[g * 16 + jl];
        float val;
        if (col < RBF_N)      val = rbf_val(d, col) * (acc[nt][r] + P.ei_b[col]);
        else if (col == RBF_N) val = d;
        else                   val = 0.0f;
        st[jl * 64 + col] = (_Float16)val;
      }
    }
    __builtin_amdgcn_wave_barrier();

    // ---- GEMM2: e(192: h_cat|filt|d|pad) @ W_eo1 -> 16 x 64
    v8f acc2[4];
#pragma unroll
    for (int q = 0; q < 4; ++q) acc2[q] = zero_v8f();
#pragma unroll
    for (int ks = 0; ks < 6; ++ks) {
      v16h a;
#pragma unroll
      for (int e = 0; e < 16; ++e) {
        int k = ks * 32 + koff + (e < 8 ? e : e + 8);
        _Float16 v;
        if (k < 64)        v = sh_j[m * 64 + k];
        else if (k < 128)  v = sh_i[g * 64 + (k - 64)];
        else               v = st[m * 64 + (k - 128)];
        a[e] = v;
      }
#pragma unroll
      for (int nt = 0; nt < 4; ++nt)
        acc2[nt] = wmma_f16(a, load_b_frag(P.eo1_p + (nt * 6 + ks) * 512, lane), acc2[nt]);
    }
    __builtin_amdgcn_wave_barrier();
#pragma unroll
    for (int nt = 0; nt < 4; ++nt) {
      int col = nt * 16 + n;
#pragma unroll
      for (int r = 0; r < 8; ++r)
        st[(mhi + r) * 64 + col] = (_Float16)silu_f(acc2[nt][r] + P.eo1_b[col]);
    }
    __builtin_amdgcn_wave_barrier();

    // ---- GEMM3: h1(64) @ W_eo2 -> h_e (16 x 64)
    v8f acc3[4];
#pragma unroll
    for (int q = 0; q < 4; ++q) acc3[q] = zero_v8f();
#pragma unroll
    for (int ks = 0; ks < 2; ++ks) {
      v16h a;
#pragma unroll
      for (int e = 0; e < 16; ++e) {
        int k = ks * 32 + koff + (e < 8 ? e : e + 8);
        a[e] = st[m * 64 + k];
      }
#pragma unroll
      for (int nt = 0; nt < 4; ++nt)
        acc3[nt] = wmma_f16(a, load_b_frag(P.eo2_p + (nt * 2 + ks) * 512, lane), acc3[nt]);
    }
    __builtin_amdgcn_wave_barrier();
#pragma unroll
    for (int nt = 0; nt < 4; ++nt) {
      int col = nt * 16 + n;
#pragma unroll
      for (int r = 0; r < 8; ++r) {
        int jl = mhi + r;
        float hv = acc3[nt][r] + P.eo2_b[col];
        he[((size_t)gi * NP + (jtile * 16 + jl)) * 64 + col] = (_Float16)hv;
        st[jl * 64 + col] = (_Float16)hv;
      }
    }
    __builtin_amdgcn_wave_barrier();

    // sem head logits + combined attention logit (one softmax == sem*euc renorm)
    if (lane < 16) {
      int jl = lane;
      int gj = jtile * 16 + jl;
      float d = sd[g * 16 + jl];
      float lg[4];
#pragma unroll
      for (int hh = 0; hh < HEADS; ++hh) {
        float s = P.sem_b[hh];
        for (int k = 0; k < 64; ++k) s += (float)st[jl * 64 + k] * P.sem_w[k * HEADS + hh];
        float c = (s > 0.0f) ? s : 2.0f * (__expf(0.5f * s) - 1.0f);   // celu(alpha=2)
        float l = c - d * __expf(P.log_gamma[hh]);
        if (gi == gj || gj >= NREAL) l = -1e9f;
        lg[hh] = l;
      }
      float* lp = &logits[((size_t)gi * NP + gj) * 4];
      lp[0] = lg[0]; lp[1] = lg[1]; lp[2] = lg[2]; lp[3] = lg[3];
    }
    __builtin_amdgcn_wave_barrier();
  }
}

// ---------------------------------------------------------------------------
// Row softmax stats (max, sum) over j for each (i, head)
// ---------------------------------------------------------------------------
__global__ __launch_bounds__(256) void row_softmax(
    const float* __restrict__ logits, float* __restrict__ rmax, float* __restrict__ rsum) {
  __shared__ float red[256];
  int i = blockIdx.x, tid = threadIdx.x;
  for (int h = 0; h < HEADS; ++h) {
    float m = -3.0e38f;
    for (int j = tid; j < NREAL; j += 256) m = fmaxf(m, logits[((size_t)i * NP + j) * 4 + h]);
    red[tid] = m; __syncthreads();
    for (int s = 128; s > 0; s >>= 1) { if (tid < s) red[tid] = fmaxf(red[tid], red[tid + s]); __syncthreads(); }
    m = red[0]; __syncthreads();
    float sum = 0.0f;
    for (int j = tid; j < NREAL; j += 256) sum += __expf(logits[((size_t)i * NP + j) * 4 + h] - m);
    red[tid] = sum; __syncthreads();
    for (int s = 128; s > 0; s >>= 1) { if (tid < s) red[tid] += red[tid + s]; __syncthreads(); }
    if (tid == 0) { rmax[i * 4 + h] = m; rsum[i * 4 + h] = red[0]; }
    __syncthreads();
  }
}

// ---------------------------------------------------------------------------
// pass3: per node i: X = h_e*att, coeff = tanh(X @ Wmix) via WMMA,
// combos/agg accumulation, then post/node/vel MLPs and x/v update.
// ---------------------------------------------------------------------------
__global__ __launch_bounds__(256) void edge_pass3(
    const _Float16* __restrict__ he, const float* __restrict__ logits,
    const float* __restrict__ rmax, const float* __restrict__ rsum,
    const float* __restrict__ x, const float* __restrict__ hcur, const float* __restrict__ vcur,
    LayerPtrs P, float* __restrict__ hnext, float* __restrict__ xnext, float* __restrict__ vnext) {
  __shared__ _Float16 X[16 * 256];
  __shared__ float unitt[16][3];
  __shared__ float combos[256 * 3];
  __shared__ float aggs[256];
  __shared__ float buf[256];
  __shared__ float tmpA[64], tmpB[64], hnew[64];
  __shared__ float dv[4];
  __shared__ float vs_sh;

  int i = blockIdx.x, tid = threadIdx.x, wave = tid >> 5, lane = tid & 31;
  for (int t = tid; t < 768; t += 256) combos[t] = 0.0f;
  if (tid < 4) dv[tid] = 0.0f;

  float rm0 = rmax[i * 4 + 0], rm1 = rmax[i * 4 + 1], rm2 = rmax[i * 4 + 2], rm3 = rmax[i * 4 + 3];
  float ri0 = 1.0f / rsum[i * 4 + 0], ri1 = 1.0f / rsum[i * 4 + 1];
  float ri2 = 1.0f / rsum[i * 4 + 2], ri3 = 1.0f / rsum[i * 4 + 3];
  float xi0 = x[i * 3 + 0], xi1 = x[i * 3 + 1], xi2 = x[i * 3 + 2];
  float aggreg = 0.0f;
  __syncthreads();

  const int mm = lane & 15, koff = (lane >> 4) * 8;
  const int nn = lane & 15, mhi = (lane >> 4) * 8;

  for (int jt = 0; jt < 17; ++jt) {
    // build X[jl][c] = h_e[i, gj, c>>2] * att[i, gj, c&3]
    {
      int jl = tid >> 4, cb = (tid & 15) << 4;
      int gj = jt * 16 + jl;
      const float* lp = &logits[((size_t)i * NP + gj) * 4];
      float a0 = __expf(lp[0] - rm0) * ri0;
      float a1 = __expf(lp[1] - rm1) * ri1;
      float a2 = __expf(lp[2] - rm2) * ri2;
      float a3 = __expf(lp[3] - rm3) * ri3;
      const _Float16* hr = he + ((size_t)i * NP + gj) * 64;
      float hv[4];
#pragma unroll
      for (int q = 0; q < 4; ++q) hv[q] = (float)hr[(cb >> 2) + q];
#pragma unroll
      for (int q = 0; q < 16; ++q) {
        float att = (q & 3) == 0 ? a0 : ((q & 3) == 1 ? a1 : ((q & 3) == 2 ? a2 : a3));
        X[jl * 256 + cb + q] = (_Float16)(hv[q >> 2] * att);
      }
    }
    if (tid < 16) {
      int gj = jt * 16 + tid;
      float dxx = x[gj * 3 + 0] - xi0, dyy = x[gj * 3 + 1] - xi1, dzz = x[gj * 3 + 2] - xi2;
      float d = sqrtf(dxx * dxx + dyy * dyy + dzz * dzz + 1e-14f);
      float inv = 1.0f / (d + 1e-5f);
      unitt[tid][0] = dxx * inv; unitt[tid][1] = dyy * inv; unitt[tid][2] = dzz * inv;
    }
    __syncthreads();

    // agg (pre-mix sum over j), thread owns c = tid
    {
      float s = 0.0f;
#pragma unroll
      for (int j2 = 0; j2 < 16; ++j2) s += (float)X[j2 * 256 + tid];
      aggreg += s;
    }

    // coeff tile: wave handles N-tiles {2w, 2w+1}; K = 256 (8 steps)
    v8f acc0 = zero_v8f(), acc1 = zero_v8f();
#pragma unroll
    for (int ks = 0; ks < 8; ++ks) {
      v16h a;
#pragma unroll
      for (int e = 0; e < 16; ++e) {
        int k = ks * 32 + koff + (e < 8 ? e : e + 8);
        a[e] = X[mm * 256 + k];
      }
      acc0 = wmma_f16(a, load_b_frag(P.mix_p + ((wave * 2 + 0) * 8 + ks) * 512, lane), acc0);
      acc1 = wmma_f16(a, load_b_frag(P.mix_p + ((wave * 2 + 1) * 8 + ks) * 512, lane), acc1);
    }
#pragma unroll
    for (int u = 0; u < 2; ++u) {
      v8f acc = (u == 0) ? acc0 : acc1;
      int c = (wave * 2 + u) * 16 + nn;
#pragma unroll
      for (int r = 0; r < 8; ++r) {
        float cf = tanhf(acc[r]);
        int j2 = mhi + r;
        atomicAdd(&combos[c * 3 + 0], unitt[j2][0] * cf);
        atomicAdd(&combos[c * 3 + 1], unitt[j2][1] * cf);
        atomicAdd(&combos[c * 3 + 2], unitt[j2][2] * cf);
      }
    }
    __syncthreads();
  }

  aggs[tid] = aggreg;
  const float inv_n = 1.0f / 257.0f;
  float cs0 = combos[tid * 3 + 0] * inv_n;
  float cs1 = combos[tid * 3 + 1] * inv_n;
  float cs2 = combos[tid * 3 + 2] * inv_n;
  buf[tid] = cs0 * cs0 + cs1 * cs1 + cs2 * cs2;      // h_comb[c]
  atomicAdd(&dv[0], cs0); atomicAdd(&dv[1], cs1); atomicAdd(&dv[2], cs2);
  __syncthreads();

  if (tid < 64) {                                     // post1: 256 -> 64
    float s = P.p1_b[tid];
    for (int c = 0; c < 256; ++c) s += buf[c] * P.p1_w[c * 64 + tid];
    tmpA[tid] = silu_f(s);
  }
  __syncthreads();
  if (tid < 64) {                                     // post2: 64 -> 64
    float s = P.p2_b[tid];
    for (int k = 0; k < 64; ++k) s += tmpA[k] * P.p2_w[k * 64 + tid];
    tmpB[tid] = silu_f(s);
  }
  __syncthreads();
  if (tid < 64) {                                     // node1: [h|agg|h_comb](384) -> 64
    float s = P.n1_b[tid];
    for (int k = 0; k < 64; ++k)  s += hcur[i * 64 + k] * P.n1_w[k * 64 + tid];
    for (int k = 0; k < 256; ++k) s += aggs[k] * P.n1_w[(64 + k) * 64 + tid];
    for (int k = 0; k < 64; ++k)  s += tmpB[k] * P.n1_w[(320 + k) * 64 + tid];
    tmpA[tid] = silu_f(s);
  }
  __syncthreads();
  if (tid < 64) {                                     // node2 + residual
    float s = P.n2_b[tid];
    for (int k = 0; k < 64; ++k) s += tmpA[k] * P.n2_w[k * 64 + tid];
    float hv = hcur[i * 64 + tid] + silu_f(s);
    hnew[tid] = hv;
    hnext[i * 64 + tid] = hv;
  }
  __syncthreads();
  if (tid < 64) {                                     // vel1
    float s = P.v1_b[tid];
    for (int k = 0; k < 64; ++k) s += hnew[k] * P.v1_w[k * 64 + tid];
    tmpA[tid] = silu_f(s);
  }
  __syncthreads();
  if (tid == 0) {                                     // vel2 (no bias)
    float s = 0.0f;
    for (int k = 0; k < 64; ++k) s += tmpA[k] * P.v2_w[k];
    vs_sh = s;
  }
  __syncthreads();
  if (tid < 3) {
    float vn = vs_sh * vcur[i * 3 + tid] + dv[tid] * (1.0f / 256.0f);
    vnext[i * 3 + tid] = vn;
    xnext[i * 3 + tid] = x[i * 3 + tid] + vn;
  }
}

// ---------------------------------------------------------------------------
// final: emb_out, scale head, translation/mean, outputs (x, v_new, log_det)
// ---------------------------------------------------------------------------
__global__ __launch_bounds__(256) void final_kernel(
    const float* __restrict__ hfin, const float* __restrict__ xfin,
    const float* __restrict__ x0, const float* __restrict__ vin,
    const float* __restrict__ eo1_b, const float* __restrict__ eo1_w,
    const float* __restrict__ eo2_b, const float* __restrict__ eo2_w,
    const float* __restrict__ s1_b, const float* __restrict__ s1_w,
    const float* __restrict__ s2_w, float* __restrict__ out) {
  __shared__ float red[256];
  __shared__ float mtr[3];
  __shared__ float scale_sh;
  int t = threadIdx.x;

  float e1[64];
  for (int o = 0; o < 64; ++o) {
    float s = eo1_b[o];
    for (int k = 0; k < 64; ++k) s += hfin[t * 64 + k] * eo1_w[k * 64 + o];
    e1[o] = silu_f(s);
  }
  float hsc = eo2_b[0];
  for (int k = 0; k < 64; ++k) hsc += e1[k] * eo2_w[k];
  float sv = 0.0f;
  for (int o = 0; o < 64; ++o) sv += silu_f(hsc * s1_w[o] + s1_b[o]) * s2_w[o];
  sv = tanhf(sv);

  red[t] = sv; __syncthreads();
  for (int s = 128; s > 0; s >>= 1) { if (t < s) red[t] += red[t + s]; __syncthreads(); }
  if (t == 0) scale_sh = red[0] / 256.0f;
  __syncthreads();

  float tr[3];
#pragma unroll
  for (int c = 0; c < 3; ++c) tr[c] = xfin[t * 3 + c] - x0[t * 3 + c];
#pragma unroll
  for (int c = 0; c < 3; ++c) {
    red[t] = tr[c]; __syncthreads();
    for (int s = 128; s > 0; s >>= 1) { if (t < s) red[t] += red[t + s]; __syncthreads(); }
    if (t == 0) mtr[c] = red[0] / 256.0f;
    __syncthreads();
  }
  float scale = scale_sh;
  float es = __expf(scale);
#pragma unroll
  for (int c = 0; c < 3; ++c) {
    out[t * 3 + c] = x0[t * 3 + c];                            // x (unchanged)
    out[768 + t * 3 + c] = es * vin[t * 3 + c] + (tr[c] - mtr[c]);  // v_new
  }
  if (t == 0) out[1536] = scale * 768.0f;                      // log_det
}

// ---------------------------------------------------------------------------
// host launcher
// ---------------------------------------------------------------------------
extern "C" void kernel_launch(void* const* d_in, const int* in_sizes, int n_in,
                              void* d_out, int out_size, void* d_ws, size_t ws_size,
                              hipStream_t stream) {
  (void)in_sizes; (void)n_in; (void)out_size; (void)ws_size;
  // Input flattening: h, x, v, then params as a jax pytree (dict keys sorted):
  // emb_in{b,w}, emb_out1{b,w}, emb_out2{b,w}, layers[0..2] x 21 leaves, scale1{b,w}, scale2{w}
  const float* in_h = (const float*)d_in[0];
  const float* in_x = (const float*)d_in[1];
  const float* in_v = (const float*)d_in[2];
  const float* emb_in_b  = (const float*)d_in[3];
  const float* emb_in_w  = (const float*)d_in[4];
  const float* emb_o1_b  = (const float*)d_in[5];
  const float* emb_o1_w  = (const float*)d_in[6];
  const float* emb_o2_b  = (const float*)d_in[7];
  const float* emb_o2_w  = (const float*)d_in[8];
  const float* s1_b = (const float*)d_in[72];
  const float* s1_w = (const float*)d_in[73];
  const float* s2_w = (const float*)d_in[74];

  char* p = (char*)d_ws;
  auto carve = [&](size_t bytes) -> char* {
    char* r = p; p += (bytes + 255) & ~(size_t)255; return r;
  };
  _Float16* pk_ei[3];  _Float16* pk_eo1[3]; _Float16* pk_eo2[3]; _Float16* pk_mix[3];
  for (int L = 0; L < 3; ++L) {
    pk_ei[L]  = (_Float16*)carve(4 * 4 * 512 * 2);     // K=128,N=64(pad)
    pk_eo1[L] = (_Float16*)carve(4 * 6 * 512 * 2);     // K=192(pad),N=64
    pk_eo2[L] = (_Float16*)carve(4 * 2 * 512 * 2);     // K=64,N=64
    pk_mix[L] = (_Float16*)carve(16 * 8 * 512 * 2);    // K=256,N=256
  }
  float* hA = (float*)carve((size_t)NP * 64 * 4);
  float* hB = (float*)carve((size_t)NP * 64 * 4);
  float* xA = (float*)carve((size_t)NP * 3 * 4);
  float* xB = (float*)carve((size_t)NP * 3 * 4);
  float* vA = (float*)carve((size_t)NP * 3 * 4);
  float* vB = (float*)carve((size_t)NP * 3 * 4);
  _Float16* h16   = (_Float16*)carve((size_t)NP * 64 * 2);
  _Float16* he    = (_Float16*)carve((size_t)NP * NP * 64 * 2);
  float* logits   = (float*)carve((size_t)NP * NP * 4 * 4);
  float* rmax     = (float*)carve((size_t)NP * 4 * 4);
  float* rsum     = (float*)carve((size_t)NP * 4 * 4);

  // ---- pack weights (per call; deterministic) ----
  for (int L = 0; L < 3; ++L) {
    int base = 9 + L * 21;
    const float* ei_w  = (const float*)d_in[base + 1];
    const float* eo1_w = (const float*)d_in[base + 3];
    const float* eo2_w = (const float*)d_in[base + 5];
    const float* mix_w = (const float*)d_in[base + 20];
    pack_weight<<<32, 256, 0, stream>>>(ei_w,  pk_ei[L],  128, 50, 4, 4);
    pack_weight<<<48, 256, 0, stream>>>(eo1_w, pk_eo1[L], 179, 64, 6, 4);
    pack_weight<<<16, 256, 0, stream>>>(eo2_w, pk_eo2[L],  64, 64, 2, 4);
    pack_weight<<<256, 256, 0, stream>>>(mix_w, pk_mix[L], 256, 256, 8, 16);
  }

  // ---- prep ----
  prep_kernel<<<NP, 64, 0, stream>>>(in_h, in_x, emb_in_b, emb_in_w, hA, xA, vA);

  float *hc = hA, *xc = xA, *vc = vA, *hn = hB, *xn = xB, *vn = vB;
  for (int L = 0; L < 3; ++L) {
    int base = 9 + L * 21;
    LayerPtrs P;
    P.ei_b  = (const float*)d_in[base + 0];
    P.eo1_b = (const float*)d_in[base + 2];
    P.eo2_b = (const float*)d_in[base + 4];
    P.log_gamma = (const float*)d_in[base + 6];
    P.n1_b = (const float*)d_in[base + 7];  P.n1_w = (const float*)d_in[base + 8];
    P.n2_b = (const float*)d_in[base + 9];  P.n2_w = (const float*)d_in[base + 10];
    P.p1_b = (const float*)d_in[base + 11]; P.p1_w = (const float*)d_in[base + 12];
    P.p2_b = (const float*)d_in[base + 13]; P.p2_w = (const float*)d_in[base + 14];
    P.sem_b = (const float*)d_in[base + 15]; P.sem_w = (const float*)d_in[base + 16];
    P.v1_b = (const float*)d_in[base + 17]; P.v1_w = (const float*)d_in[base + 18];
    P.v2_w = (const float*)d_in[base + 19];
    P.ei_p = pk_ei[L]; P.eo1_p = pk_eo1[L]; P.eo2_p = pk_eo2[L]; P.mix_p = pk_mix[L];

    h_to_f16<<<(NP * 64 + 255) / 256, 256, 0, stream>>>(hc, h16);
    edge_pass1<<<dim3(17, 17), 256, 0, stream>>>(h16, xc, P, he, logits);
    row_softmax<<<NREAL, 256, 0, stream>>>(logits, rmax, rsum);
    edge_pass3<<<NREAL, 256, 0, stream>>>(he, logits, rmax, rsum, xc, hc, vc,
                                          P, hn, xn, vn);
    float* t;
    t = hc; hc = hn; hn = t;
    t = xc; xc = xn; xn = t;
    t = vc; vc = vn; vn = t;
  }

  final_kernel<<<1, 256, 0, stream>>>(hc, xc, in_x, in_v,
                                      emb_o1_b, emb_o1_w, emb_o2_b, emb_o2_w,
                                      s1_b, s1_w, s2_w, (float*)d_out);
}
